// LlamaAttention_17085379903943
// MI455X (gfx1250) — matrix operand order repaired and checked by reading, not compiled, not run on hardware
//
#include <hip/hip_runtime.h>
#include <hip/hip_bf16.h>
#include <math.h>

// ---------------------------------------------------------------------------
// Types for CDNA5 WMMA (wave32): 16x16x32 bf16 -> f32
// ---------------------------------------------------------------------------
typedef __attribute__((ext_vector_type(16))) __bf16 bf16x16;
typedef __attribute__((ext_vector_type(8)))  __bf16 bf16x8;
typedef __attribute__((ext_vector_type(8)))  float  floatx8;

#define WMMA_BF16(a, b, c) \
  __builtin_amdgcn_wmma_f32_16x16x32_bf16(false, (a), false, (b), (short)0, (c), false, false)

// A-matrix 16x32 bf16 fragment: lanes 0-15 hold K {0..15}, lanes 16-31 hold K {16..31}
// (contiguous, matching B). Caller passes kb = (lane>>4)*8.
static __device__ __forceinline__ bf16x16 a_frag_bf16(const __bf16* rowptr, int kb) {
  union { bf16x16 v; bf16x8 h[2]; } u;
  u.h[0] = *(const bf16x8*)(rowptr + kb * 2);
  u.h[1] = *(const bf16x8*)(rowptr + kb * 2 + 8);
  return u.v;
}

// ---------------------------------------------------------------------------
// f32 -> bf16 conversion (grid-stride)
// ---------------------------------------------------------------------------
__global__ __launch_bounds__(256) void convert_bf16_kernel(
    const float* __restrict__ src, __bf16* __restrict__ dst, int n) {
  for (int i = blockIdx.x * blockDim.x + threadIdx.x; i < n; i += gridDim.x * blockDim.x)
    dst[i] = (__bf16)src[i];
}

// ---------------------------------------------------------------------------
// RoPE tables: cost/sint[s*64 + f] = cos/sin(s * 10000^(-f/64)), f in [0,64)
// ---------------------------------------------------------------------------
__global__ __launch_bounds__(256) void rope_tables_kernel(
    float* __restrict__ cost, float* __restrict__ sint, int S) {
  int idx = blockIdx.x * blockDim.x + threadIdx.x;
  if (idx >= S * 64) return;
  int s = idx >> 6, f = idx & 63;
  float inv = __expf(-(float)f * (9.210340371976184f / 64.0f)); // ln(10000)=9.2103...
  float ang = (float)s * inv;
  cost[idx] = __cosf(ang);
  sint[idx] = __sinf(ang);
}

// ---------------------------------------------------------------------------
// LoRA mid projection: L[m][r] = sum_k X[m,k] * A[r,k]   (rank 16, tiny FLOPs)
// ---------------------------------------------------------------------------
__global__ __launch_bounds__(256) void lora_mid_kernel(
    const float* __restrict__ X, const float* __restrict__ Aq,
    const float* __restrict__ Av, __bf16* __restrict__ Lq,
    __bf16* __restrict__ Lv, int M, int K) {
  int idx = blockIdx.x * blockDim.x + threadIdx.x;
  if (idx >= M * 32) return;
  int m = idx >> 5;
  int rem = idx & 31;
  int which = rem >> 4;
  int r = rem & 15;
  const float* A = which ? Av : Aq;
  const float* xr = X + (size_t)m * K;
  const float* ar = A + (size_t)r * K;
  float acc = 0.f;
  for (int k = 0; k < K; k += 4) {
    acc += xr[k] * ar[k] + xr[k + 1] * ar[k + 1] +
           xr[k + 2] * ar[k + 2] + xr[k + 3] * ar[k + 3];
  }
  (which ? Lv : Lq)[(size_t)m * 16 + r] = (__bf16)acc;
}

// ---------------------------------------------------------------------------
// Tiled WMMA GEMM: C[M,N] = A[M,K](bf16) @ Bw[N,K]^T (bf16)
//                  (+ lora_scale * L[M,16] @ Bl[N,16]^T), output f32 or bf16.
// Block tile 128x128, 8 waves; wave (wm,wn) computes 64x32 = 4x2 WMMA tiles.
// ---------------------------------------------------------------------------
__global__ __launch_bounds__(256) void gemm_bf16_wmma_kernel(
    const __bf16* __restrict__ A, const __bf16* __restrict__ Bw,
    const __bf16* __restrict__ L, const __bf16* __restrict__ Bl,
    float lora_scale, __bf16* __restrict__ Cbf, float* __restrict__ Cf,
    int M, int N, int K) {
  const int tid = threadIdx.x;
  const int lane = tid & 31;
  const int wave = tid >> 5;
  const int wm = wave & 1;   // 0..1 : 64-row strip
  const int wn = wave >> 1;  // 0..3 : 32-col strip
  const int half = lane >> 4;
  const int l15 = lane & 15;
  const int rowBase = blockIdx.x * 128 + wm * 64;
  const int colBase = blockIdx.y * 128 + wn * 32;

  floatx8 acc[4][2];
#pragma unroll
  for (int mt = 0; mt < 4; ++mt)
#pragma unroll
    for (int nt = 0; nt < 2; ++nt) acc[mt][nt] = (floatx8){};

  for (int k0 = 0; k0 < K; k0 += 32) {
    bf16x16 bfr[2];
#pragma unroll
    for (int nt = 0; nt < 2; ++nt) {
      int n = colBase + nt * 16 + l15;
      bfr[nt] = *(const bf16x16*)(Bw + (size_t)n * K + k0 + half * 16);
    }
#pragma unroll
    for (int mt = 0; mt < 4; ++mt) {
      int m = rowBase + mt * 16 + l15;
      const __bf16* arow = A + (size_t)m * K + k0;
      __builtin_prefetch(arow + 32, 0, 3);
      bf16x16 afr = a_frag_bf16(arow, half * 8);
#pragma unroll
      for (int nt = 0; nt < 2; ++nt)
        acc[mt][nt] = WMMA_BF16(afr, bfr[nt], acc[mt][nt]);
    }
  }

  // Epilogue: C layout (ISA 7.12.2): element v of frag -> M = v + 8*half, N = l15.
  const bool hasLora = (L != nullptr);
#pragma unroll
  for (int mt = 0; mt < 4; ++mt) {
#pragma unroll
    for (int nt = 0; nt < 2; ++nt) {
      int n = colBase + nt * 16 + l15;
#pragma unroll
      for (int v = 0; v < 8; ++v) {
        int m = rowBase + mt * 16 + v + 8 * half;
        float c = acc[mt][nt][v];
        if (hasLora) {
          const __bf16* lm = L + (size_t)m * 16;
          const __bf16* bn = Bl + (size_t)n * 16;
          float d = 0.f;
#pragma unroll
          for (int r = 0; r < 16; ++r) d += (float)lm[r] * (float)bn[r];
          c += lora_scale * d;
        }
        if (Cbf) Cbf[(size_t)m * N + n] = (__bf16)c;
        else     Cf[(size_t)m * N + n] = c;
      }
    }
  }
}

// ---------------------------------------------------------------------------
// RoPE + relayout: T[m][n] (m = b*S+s, n = h*128+d) -> out[(b*16+h)*S + s][d]
// q' = q*cos + rotate_half(q)*sin  ; pairs d <-> d^64
// ---------------------------------------------------------------------------
__global__ __launch_bounds__(256) void rope_relayout_kernel(
    const __bf16* __restrict__ T, const float* __restrict__ cost,
    const float* __restrict__ sint, __bf16* __restrict__ out, int S) {
  int idx = blockIdx.x * blockDim.x + threadIdx.x;
  if (idx >= (S * 2) * 2048) return;  // M=2S rows x H=2048 cols
  int m = idx >> 11;
  int n = idx & 2047;
  int b = m / S, s = m % S;
  int h = n >> 7, d = n & 127;
  float x  = (float)T[idx];
  float xp = (float)T[(size_t)m * 2048 + (n - d) + (d ^ 64)];
  float c  = cost[s * 64 + (d & 63)];
  float sn = sint[s * 64 + (d & 63)];
  float rot = (d < 64) ? -xp : xp;
  float y = x * c + rot * sn;
  out[(((size_t)(b * 16 + h)) * S + s) * 128 + d] = (__bf16)y;
}

// V relayout: T[m][n] -> Vt[(b*16+h)*128 + d][s]  (transposed for P@V B-frags)
__global__ __launch_bounds__(256) void v_relayout_kernel(
    const __bf16* __restrict__ T, __bf16* __restrict__ Vt, int S) {
  int idx = blockIdx.x * blockDim.x + threadIdx.x;
  if (idx >= (S * 2) * 2048) return;
  int m = idx >> 11;
  int n = idx & 2047;
  int b = m / S, s = m % S;
  int h = n >> 7, d = n & 127;
  Vt[(((size_t)(b * 16 + h)) * 128 + d) * S + s] = T[idx];
}

// ---------------------------------------------------------------------------
// Flash attention: per (b,h) x 64-row Q block. Online softmax, WMMA for
// Q@K^T and P@V. Q in [bh][s][128] bf16 (A-frags held in registers for all
// 16 key tiles), K same layout, Vt in [bh][d][S] bf16. Probs stored bf16 in
// LDS so P-fragments are straight 16B ds loads (no converts in WMMA loop).
// ctx written bf16 in [b][s][h*128+d] layout for the output projection.
// ---------------------------------------------------------------------------
__global__ __launch_bounds__(256) void flash_attn_kernel(
    const __bf16* __restrict__ Qg, const __bf16* __restrict__ Kg,
    const __bf16* __restrict__ Vt, __bf16* __restrict__ ctx, int S) {
  __shared__ __align__(32) float  Ss[64 * 128];  // f32 scores
  __shared__ __align__(32) __bf16 Ps[64 * 128];  // bf16 probs
  __shared__ float rowm[64], rowsum[64], alf[64], mnew[64];
  __shared__ float pmax[128], psum[128];

  const int bh = blockIdx.x;  // 0..31
  const int qb = blockIdx.y;  // 0..S/64-1
  const int tid = threadIdx.x;
  const int lane = tid & 31;
  const int wave = tid >> 5;  // 8 waves; wave w owns 16 key-cols / 16 dims
  const int half = lane >> 4;
  const int l15 = lane & 15;

  const __bf16* Qbase = Qg + ((size_t)bh * S + qb * 64) * 128;

  // Hoist Q A-fragments into registers: reused for every key tile.
  bf16x16 qfr[4][4];
#pragma unroll
  for (int mt = 0; mt < 4; ++mt)
#pragma unroll
    for (int ksi = 0; ksi < 4; ++ksi)
      qfr[mt][ksi] = a_frag_bf16(Qbase + (size_t)(mt * 16 + l15) * 128 + ksi * 32, half * 8);

  if (tid < 64) { rowm[tid] = -1e30f; rowsum[tid] = 0.f; }
  __syncthreads();

  floatx8 cacc[4];
#pragma unroll
  for (int mt = 0; mt < 4; ++mt) cacc[mt] = (floatx8){};

  const float scale = 0.08838834764831845f;  // 1/sqrt(128)

  for (int j = 0; j < S; j += 128) {
    // ---- scores: S_tile[64 x 16] per wave = Q(64x128) @ K_cols(128x16)^T
    floatx8 sacc[4];
#pragma unroll
    for (int mt = 0; mt < 4; ++mt) sacc[mt] = (floatx8){};
    {
      int key = j + wave * 16 + l15;
      const __bf16* Krow = Kg + ((size_t)bh * S + key) * 128;
#pragma unroll
      for (int ksi = 0; ksi < 4; ++ksi) {
        bf16x16 bfr = *(const bf16x16*)(Krow + ksi * 32 + half * 16);
#pragma unroll
        for (int mt = 0; mt < 4; ++mt)
          sacc[mt] = WMMA_BF16(qfr[mt][ksi], bfr, sacc[mt]);
      }
    }
#pragma unroll
    for (int mt = 0; mt < 4; ++mt)
#pragma unroll
      for (int v = 0; v < 8; ++v) {
        int r = mt * 16 + v + 8 * half;
        Ss[r * 128 + wave * 16 + l15] = sacc[mt][v] * scale;
      }
    __syncthreads();

    // ---- online softmax (two 64-col segments per row, 128 worker threads)
    if (tid < 128) {
      int r = tid & 63, seg = tid >> 6;
      const float* row = Ss + r * 128 + seg * 64;
      float mb = -1e30f;
      for (int c = 0; c < 64; ++c) mb = fmaxf(mb, row[c]);
      pmax[seg * 64 + r] = mb;
    }
    __syncthreads();
    if (tid < 64) {
      float mo = rowm[tid];
      float mn = fmaxf(mo, fmaxf(pmax[tid], pmax[64 + tid]));
      mnew[tid] = mn;
      alf[tid] = __expf(mo - mn);
      rowm[tid] = mn;
    }
    __syncthreads();
    if (tid < 128) {
      int r = tid & 63, seg = tid >> 6;
      float mn = mnew[r];
      const float* row = Ss + r * 128 + seg * 64;
      __bf16* prow = Ps + r * 128 + seg * 64;
      float se = 0.f;
      for (int c = 0; c < 64; ++c) {
        float e = __expf(row[c] - mn);
        prow[c] = (__bf16)e;
        se += e;
      }
      psum[seg * 64 + r] = se;
    }
    __syncthreads();
    if (tid < 64) rowsum[tid] = rowsum[tid] * alf[tid] + psum[tid] + psum[64 + tid];

    // ---- rescale running context by alpha(row), then ctx += P @ V_block
#pragma unroll
    for (int mt = 0; mt < 4; ++mt)
#pragma unroll
      for (int v = 0; v < 8; ++v)
        cacc[mt][v] *= alf[mt * 16 + v + 8 * half];

    {
      int d = wave * 16 + l15;
      const __bf16* Vrow = Vt + ((size_t)bh * 128 + d) * S + j;
#pragma unroll
      for (int ksi = 0; ksi < 4; ++ksi) {
        bf16x16 bfr = *(const bf16x16*)(Vrow + ksi * 32 + half * 16);
#pragma unroll
        for (int mt = 0; mt < 4; ++mt) {
          bf16x16 afr = a_frag_bf16(Ps + (mt * 16 + l15) * 128 + ksi * 32, half * 8);
          cacc[mt] = WMMA_BF16(afr, bfr, cacc[mt]);
        }
      }
    }
    __syncthreads();  // Ss/Ps reused next iteration
  }

  // ---- normalize and store ctx in [b][s][h*128+d] layout (bf16)
  const int b = bh >> 4, h = bh & 15;
  const int d = wave * 16 + l15;
#pragma unroll
  for (int mt = 0; mt < 4; ++mt)
#pragma unroll
    for (int v = 0; v < 8; ++v) {
      int r = mt * 16 + v + 8 * half;
      float val = cacc[mt][v] / rowsum[r];
      int srow = qb * 64 + r;
      ctx[((size_t)b * S + srow) * 2048 + h * 128 + d] = (__bf16)val;
    }
}

// ---------------------------------------------------------------------------
// Host-side launch
// ---------------------------------------------------------------------------
extern "C" void kernel_launch(void* const* d_in, const int* in_sizes, int n_in,
                              void* d_out, int out_size, void* d_ws, size_t ws_size,
                              hipStream_t stream) {
  (void)in_sizes; (void)n_in; (void)out_size; (void)ws_size;
  const int Bsz = 2, S = 2048, H = 2048, R = 16;
  const int M = Bsz * S;  // 4096

  const float* hs = (const float*)d_in[0];
  const float* Wq = (const float*)d_in[1];
  const float* Wk = (const float*)d_in[2];
  const float* Wv = (const float*)d_in[3];
  const float* Wo = (const float*)d_in[4];
  const float* Aq = (const float*)d_in[5];
  const float* Bq = (const float*)d_in[6];
  const float* Av = (const float*)d_in[7];
  const float* Bv = (const float*)d_in[8];
  // d_in[9] = position_ids = arange(S): positions are s directly.
  float* out = (float*)d_out;

  // Workspace carve-up (256B aligned)
  size_t off = 0;
  auto wsa = [&](size_t bytes) -> char* {
    char* p = (char*)d_ws + off;
    off += (bytes + 255) & ~(size_t)255;
    return p;
  };
  __bf16* Xbf   = (__bf16*)wsa((size_t)M * H * 2);
  __bf16* Wqbf  = (__bf16*)wsa((size_t)H * H * 2);
  __bf16* Wkbf  = (__bf16*)wsa((size_t)H * H * 2);
  __bf16* Wvbf  = (__bf16*)wsa((size_t)H * H * 2);
  __bf16* Wobf  = (__bf16*)wsa((size_t)H * H * 2);
  __bf16* Bqbf  = (__bf16*)wsa((size_t)H * R * 2);
  __bf16* Bvbf  = (__bf16*)wsa((size_t)H * R * 2);
  __bf16* Lq    = (__bf16*)wsa((size_t)M * R * 2);
  __bf16* Lv    = (__bf16*)wsa((size_t)M * R * 2);
  float*  cost  = (float*)wsa((size_t)S * 64 * 4);
  float*  sint  = (float*)wsa((size_t)S * 64 * 4);
  __bf16* tmp   = (__bf16*)wsa((size_t)M * H * 2);  // reused for Q,K,V raw proj
  __bf16* Qbf   = (__bf16*)wsa((size_t)M * H * 2);  // [bh][s][128]
  __bf16* Kbf   = (__bf16*)wsa((size_t)M * H * 2);  // [bh][s][128]
  __bf16* Vt    = (__bf16*)wsa((size_t)M * H * 2);  // [bh][d][S]
  __bf16* ctxbf = (__bf16*)wsa((size_t)M * H * 2);  // [b][s][h*128+d]

  auto cgrid = [](int n) { return dim3((unsigned)((n + 255) / 256)); };

  // 1) bf16 conversions
  convert_bf16_kernel<<<cgrid(M * H), 256, 0, stream>>>(hs, Xbf, M * H);
  convert_bf16_kernel<<<cgrid(H * H), 256, 0, stream>>>(Wq, Wqbf, H * H);
  convert_bf16_kernel<<<cgrid(H * H), 256, 0, stream>>>(Wk, Wkbf, H * H);
  convert_bf16_kernel<<<cgrid(H * H), 256, 0, stream>>>(Wv, Wvbf, H * H);
  convert_bf16_kernel<<<cgrid(H * H), 256, 0, stream>>>(Wo, Wobf, H * H);
  convert_bf16_kernel<<<cgrid(H * R), 256, 0, stream>>>(Bq, Bqbf, H * R);
  convert_bf16_kernel<<<cgrid(H * R), 256, 0, stream>>>(Bv, Bvbf, H * R);

  // 2) RoPE tables, LoRA mid projections
  rope_tables_kernel<<<cgrid(S * 64), 256, 0, stream>>>(cost, sint, S);
  lora_mid_kernel<<<cgrid(M * 32), 256, 0, stream>>>(hs, Aq, Av, Lq, Lv, M, H);

  const dim3 ggrid(M / 128, H / 128);  // (32,16)

  // 3) Q projection (+LoRA) -> RoPE -> [bh][s][d]
  gemm_bf16_wmma_kernel<<<ggrid, 256, 0, stream>>>(
      Xbf, Wqbf, Lq, Bqbf, 2.0f, tmp, nullptr, M, H, H);
  rope_relayout_kernel<<<cgrid(M * H), 256, 0, stream>>>(tmp, cost, sint, Qbf, S);

  // 4) K projection -> RoPE -> [bh][s][d]
  gemm_bf16_wmma_kernel<<<ggrid, 256, 0, stream>>>(
      Xbf, Wkbf, nullptr, nullptr, 0.0f, tmp, nullptr, M, H, H);
  rope_relayout_kernel<<<cgrid(M * H), 256, 0, stream>>>(tmp, cost, sint, Kbf, S);

  // 5) V projection (+LoRA) -> transpose -> [bh][d][s]
  gemm_bf16_wmma_kernel<<<ggrid, 256, 0, stream>>>(
      Xbf, Wvbf, Lv, Bvbf, 2.0f, tmp, nullptr, M, H, H);
  v_relayout_kernel<<<cgrid(M * H), 256, 0, stream>>>(tmp, Vt, S);

  // 6) Flash attention -> ctx bf16 [b][s][H]
  flash_attn_kernel<<<dim3(Bsz * 16, S / 64), 256, 0, stream>>>(Qbf, Kbf, Vt, ctxbf, S);

  // 7) Output projection: out = ctx @ Wo^T (f32 out)
  gemm_bf16_wmma_kernel<<<ggrid, 256, 0, stream>>>(
      ctxbf, Wobf, nullptr, nullptr, 0.0f, nullptr, out, M, H, H);
}
